// latentCorrelation_4054449127560
// MI455X (gfx1250) — matrix operand order repaired and checked by reading, compiled, not run on hardware
//
#include <hip/hip_runtime.h>
#include <math.h>

typedef __attribute__((ext_vector_type(16))) _Float16 v16h;
typedef __attribute__((ext_vector_type(8)))  _Float16 v8h;
typedef __attribute__((ext_vector_type(8)))  float    v8f;

#define GRU_H 256
#define QKD   128
#define NBAT  32
#define NT    128
#define NS    64
#define BS    2048          // NBAT*NS
#define HSTRIDE 264         // h tile: 256 halfs + 8 pad
#define WSTRIDE 264         // n-gate weight tile: 256 halfs + 8 pad

union AB {
    v16h v;
    v8h  h[2];
};

// ---------------------------------------------------------------------------
// f32 -> f16 conversion for weight matrices
// ---------------------------------------------------------------------------
__global__ void cvt_f16_kernel(const float* __restrict__ src,
                               _Float16* __restrict__ dst, int n) {
    int i = blockIdx.x * blockDim.x + threadIdx.x;
    if (i < n) dst[i] = (_Float16)src[i];
}

// ---------------------------------------------------------------------------
// GRU: 128 blocks x 256 threads (8 waves). Block = one 16-row M-tile of the
// [2048,256] hidden state, resident across all 128 timesteps.
// Wave w owns columns [32w,32w+32) of each gate.
//   r/z gate weights: register-resident B fragments (32 frags = 256 VGPRs)
//   n   gate weights: staged once in LDS (128KB), streamed per step
// ---------------------------------------------------------------------------
__global__ __launch_bounds__(256, 2)
void gru_kernel(const float* __restrict__ x,        // [32,128,64]
                const float* __restrict__ w_ih,     // [768]
                const float* __restrict__ b_ih,     // [768]
                const float* __restrict__ b_hh,     // [768]
                const _Float16* __restrict__ w_hh,  // [768,256] f16
                _Float16* __restrict__ enc)         // [2048,256] f16 out
{
    __shared__ __align__(32) _Float16 wn_sh[256 * WSTRIDE];  // n-gate weights
    __shared__ __align__(32) _Float16 h_sh[16 * HSTRIDE];    // hidden tile
    __shared__ float x_sh[16];

    const int tid  = threadIdx.x;
    const int wave = tid >> 5;
    const int lane = tid & 31;
    const int l16  = lane & 15;
    const bool hiL = lane >= 16;

    const int row0  = blockIdx.x * 16;                 // global rows [row0,row0+16)
    const int bidx  = row0 >> 6;                       // batch index
    const int s0    = row0 & 63;
    const long xbase = (long)bidx * NT * NS + s0;

    const int cbase = wave * 32;                       // per-gate column slice

    // --- stage n-gate weights (w_hh rows 512..767) into LDS, one row/thread ---
    {
        const _Float16* src = w_hh + (long)(512 + tid) * GRU_H;
        _Float16* dst = &wn_sh[tid * WSTRIDE];
#pragma unroll
        for (int k = 0; k < GRU_H; k += 8)
            *(v8h*)(dst + k) = *(const v8h*)(src + k);
    }

    // --- r/z gate B fragments in registers: B[k][n] = w_hh[nglob][k] ---
    AB bfrag[2][2][8];
#pragma unroll
    for (int g = 0; g < 2; ++g)
#pragma unroll
        for (int j = 0; j < 2; ++j) {
            const int nglob = g * GRU_H + cbase + j * 16 + l16;
            const _Float16* wrow = w_hh + (long)nglob * GRU_H;
#pragma unroll
            for (int kt = 0; kt < 8; ++kt) {
                const int k0 = kt * 32 + (hiL ? 16 : 0);
                bfrag[g][j][kt].v = *(const v16h*)(wrow + k0);
            }
        }

    // --- per-lane scalar params (column-fixed) ---
    float wih[3][2], bih[3][2], bhh[3][2];
#pragma unroll
    for (int g = 0; g < 3; ++g)
#pragma unroll
        for (int j = 0; j < 2; ++j) {
            const int idx = g * GRU_H + cbase + j * 16 + l16;
            wih[g][j] = w_ih[idx];
            bih[g][j] = b_ih[idx];
            bhh[g][j] = b_hh[idx];
        }

    // --- init hidden state ---
    v8f hc[2];
#pragma unroll
    for (int j = 0; j < 2; ++j)
#pragma unroll
        for (int i = 0; i < 8; ++i) hc[j][i] = 0.0f;
    for (int i = tid; i < 16 * HSTRIDE; i += 256) h_sh[i] = (_Float16)0.0f;
    __syncthreads();

    const int rowoff = hiL ? 8 : 0;           // C-layout row offset
    const int arow_base = l16 * HSTRIDE;      // A fragment LDS row base
    const int nrow[2] = { (cbase + l16) * WSTRIDE,
                          (cbase + 16 + l16) * WSTRIDE };

    for (int t = 0; t < NT; ++t) {
        if (tid < 16) x_sh[tid] = x[xbase + (long)t * NS + tid];

        // accumulators start at b_hh (gh = h @ w_hh^T + b_hh)
        v8f acc[3][2];
#pragma unroll
        for (int g = 0; g < 3; ++g)
#pragma unroll
            for (int j = 0; j < 2; ++j)
#pragma unroll
                for (int i = 0; i < 8; ++i) acc[g][j][i] = bhh[g][j];

#pragma unroll
        for (int kt = 0; kt < 8; ++kt) {
            AB a;
            const int koff = kt * 32 + (hiL ? 16 : 0);
            a.h[0] = *(const v8h*)(&h_sh[arow_base + koff]);
            a.h[1] = *(const v8h*)(&h_sh[arow_base + koff + 8]);

            // r and z gates from register-resident B
#pragma unroll
            for (int g = 0; g < 2; ++g)
#pragma unroll
                for (int j = 0; j < 2; ++j)
                    acc[g][j] = __builtin_amdgcn_wmma_f32_16x16x32_f16(
                        false, a.v, false, bfrag[g][j][kt].v,
                        (short)0, acc[g][j], false, false);

            // n gate: stream B fragments from LDS
            const int kk = kt * 32 + (hiL ? 16 : 0);
#pragma unroll
            for (int j = 0; j < 2; ++j) {
                AB bn;
                bn.h[0] = *(const v8h*)(&wn_sh[nrow[j] + kk]);
                bn.h[1] = *(const v8h*)(&wn_sh[nrow[j] + kk + 8]);
                acc[2][j] = __builtin_amdgcn_wmma_f32_16x16x32_f16(
                    false, a.v, false, bn.v, (short)0, acc[2][j], false, false);
            }
        }
        __syncthreads();   // all waves finished reading h_sh

        float xv[8];
#pragma unroll
        for (int i = 0; i < 8; ++i) xv[i] = x_sh[rowoff + i];

#pragma unroll
        for (int j = 0; j < 2; ++j) {
            v8f hnew;
#pragma unroll
            for (int i = 0; i < 8; ++i) {
                const float xi = xv[i];
                const float r = 1.0f / (1.0f + __expf(-(xi * wih[0][j] + bih[0][j] + acc[0][j][i])));
                const float z = 1.0f / (1.0f + __expf(-(xi * wih[1][j] + bih[1][j] + acc[1][j][i])));
                const float np = xi * wih[2][j] + bih[2][j] + r * acc[2][j][i];
                const float e2 = __expf(-2.0f * np);
                const float n  = (1.0f - e2) / (1.0f + e2);   // tanh(np)
                hnew[i] = (1.0f - z) * n + z * hc[j][i];
            }
            hc[j] = hnew;
            const int col = cbase + j * 16 + l16;
#pragma unroll
            for (int i = 0; i < 8; ++i)
                h_sh[(rowoff + i) * HSTRIDE + col] = (_Float16)hnew[i];
        }
        __syncthreads();   // new h visible before next step's A loads
    }

    // final hidden state -> enc (f16)
#pragma unroll
    for (int j = 0; j < 2; ++j) {
        const int col = cbase + j * 16 + l16;
#pragma unroll
        for (int i = 0; i < 8; ++i)
            enc[(long)(row0 + rowoff + i) * GRU_H + col] = (_Float16)hc[j][i];
    }
}

// ---------------------------------------------------------------------------
// Q/K projections: Qm = enc @ wq^T + b, Km = enc @ wk^T + b. 128 blocks x
// 256 threads; wave w produces output columns [16w,16w+16) of both Q and K.
// ---------------------------------------------------------------------------
__global__ __launch_bounds__(256, 2)
void proj_kernel(const _Float16* __restrict__ enc,  // [2048,256]
                 const _Float16* __restrict__ wq,   // [128,256]
                 const float* __restrict__ wq_b,
                 const _Float16* __restrict__ wk,   // [128,256]
                 const float* __restrict__ wk_b,
                 _Float16* __restrict__ qo,         // [2048,128]
                 _Float16* __restrict__ ko)
{
    const int tid = threadIdx.x, wave = tid >> 5, lane = tid & 31, l16 = lane & 15;
    const bool hiL = lane >= 16;
    const int row0 = blockIdx.x * 16;
    const int ncol = wave * 16 + l16;

    AB bq[8], bk[8];
    {
        const _Float16* qrow = wq + (long)ncol * GRU_H;
        const _Float16* krow = wk + (long)ncol * GRU_H;
#pragma unroll
        for (int kt = 0; kt < 8; ++kt) {
            const int k0 = kt * 32 + (hiL ? 16 : 0);
            bq[kt].v = *(const v16h*)(qrow + k0);
            bk[kt].v = *(const v16h*)(krow + k0);
        }
    }
    v8f accq, acck;
    const float qb = wq_b[ncol], kb = wk_b[ncol];
#pragma unroll
    for (int i = 0; i < 8; ++i) { accq[i] = qb; acck[i] = kb; }

#pragma unroll
    for (int kt = 0; kt < 8; ++kt) {
        AB a;
        const _Float16* ap = enc + (long)(row0 + l16) * GRU_H + kt * 32 + (hiL ? 16 : 0);
        a.h[0] = *(const v8h*)(ap);
        a.h[1] = *(const v8h*)(ap + 8);
        accq = __builtin_amdgcn_wmma_f32_16x16x32_f16(false, a.v, false, bq[kt].v,
                                                      (short)0, accq, false, false);
        acck = __builtin_amdgcn_wmma_f32_16x16x32_f16(false, a.v, false, bk[kt].v,
                                                      (short)0, acck, false, false);
    }
    const int rowoff = hiL ? 8 : 0;
#pragma unroll
    for (int i = 0; i < 8; ++i) {
        qo[(long)(row0 + rowoff + i) * QKD + ncol] = (_Float16)accq[i];
        ko[(long)(row0 + rowoff + i) * QKD + ncol] = (_Float16)acck[i];
    }
}

// ---------------------------------------------------------------------------
// Per-batch scores + softmax: W = softmax(Q K^T / sqrt(128)). 32 blocks x
// 256 threads. Wave w: M-tile w/2, N-tiles {2*(w&1), 2*(w&1)+1}.
// ---------------------------------------------------------------------------
__global__ __launch_bounds__(256, 2)
void attn_kernel(const _Float16* __restrict__ qm,  // [2048,128]
                 const _Float16* __restrict__ km,  // [2048,128]
                 float* __restrict__ out)          // [32,64,64]
{
    __shared__ float sc[64][65];
    const int tid = threadIdx.x, wave = tid >> 5, lane = tid & 31, l16 = lane & 15;
    const bool hiL = lane >= 16;
    const int b   = blockIdx.x;
    const int mt  = wave >> 1;
    const int nt0 = (wave & 1) * 2;

    AB bfr[2][4];
#pragma unroll
    for (int j = 0; j < 2; ++j) {
        const _Float16* krow = km + (long)(b * 64 + (nt0 + j) * 16 + l16) * QKD;
#pragma unroll
        for (int kt = 0; kt < 4; ++kt)
            bfr[j][kt].v = *(const v16h*)(krow + kt * 32 + (hiL ? 16 : 0));
    }
    v8f acc[2];
#pragma unroll
    for (int j = 0; j < 2; ++j)
#pragma unroll
        for (int i = 0; i < 8; ++i) acc[j][i] = 0.0f;

    const int arowg = b * 64 + mt * 16 + l16;
#pragma unroll
    for (int kt = 0; kt < 4; ++kt) {
        AB a;
        const _Float16* ap = qm + (long)arowg * QKD + kt * 32 + (hiL ? 16 : 0);
        a.h[0] = *(const v8h*)(ap);
        a.h[1] = *(const v8h*)(ap + 8);
#pragma unroll
        for (int j = 0; j < 2; ++j)
            acc[j] = __builtin_amdgcn_wmma_f32_16x16x32_f16(false, a.v, false, bfr[j][kt].v,
                                                            (short)0, acc[j], false, false);
    }
    const int rowoff = hiL ? 8 : 0;
    const float scale = 0.08838834764831845f;  // 1/sqrt(128)
#pragma unroll
    for (int j = 0; j < 2; ++j) {
        const int col = (nt0 + j) * 16 + l16;
#pragma unroll
        for (int i = 0; i < 8; ++i)
            sc[mt * 16 + rowoff + i][col] = acc[j][i] * scale;
    }
    __syncthreads();

    if (tid < 64) {
        float m = -1e30f;
        for (int n = 0; n < 64; ++n) m = fmaxf(m, sc[tid][n]);
        float s = 0.0f;
        for (int n = 0; n < 64; ++n) s += __expf(sc[tid][n] - m);
        const float inv = 1.0f / s;
        float* orow = out + ((long)b * 64 + tid) * 64;
        for (int n = 0; n < 64; ++n) orow[n] = __expf(sc[tid][n] - m) * inv;
    }
}

// ---------------------------------------------------------------------------
extern "C" void kernel_launch(void* const* d_in, const int* in_sizes, int n_in,
                              void* d_out, int out_size, void* d_ws, size_t ws_size,
                              hipStream_t stream) {
    const float* x    = (const float*)d_in[0];
    const float* w_ih = (const float*)d_in[1];
    const float* w_hh = (const float*)d_in[2];
    const float* b_ih = (const float*)d_in[3];
    const float* b_hh = (const float*)d_in[4];
    const float* wq_w = (const float*)d_in[5];
    const float* wq_b = (const float*)d_in[6];
    const float* wk_w = (const float*)d_in[7];
    const float* wk_b = (const float*)d_in[8];
    float* out = (float*)d_out;

    // workspace layout (halfs)
    _Float16* ws16  = (_Float16*)d_ws;
    _Float16* whh16 = ws16;                    // 768*256   = 196608
    _Float16* wq16  = whh16 + 196608;          // 128*256   = 32768
    _Float16* wk16  = wq16 + 32768;            // 128*256   = 32768
    _Float16* enc16 = wk16 + 32768;            // 2048*256  = 524288
    _Float16* q16   = enc16 + 524288;          // 2048*128  = 262144
    _Float16* k16   = q16 + 262144;            // 2048*128  = 262144

    cvt_f16_kernel<<<(196608 + 255) / 256, 256, 0, stream>>>(w_hh, whh16, 196608);
    cvt_f16_kernel<<<(32768 + 255) / 256, 256, 0, stream>>>(wq_w, wq16, 32768);
    cvt_f16_kernel<<<(32768 + 255) / 256, 256, 0, stream>>>(wk_w, wk16, 32768);

    gru_kernel<<<BS / 16, 256, 0, stream>>>(x, w_ih, b_ih, b_hh, whh16, enc16);
    proj_kernel<<<BS / 16, 256, 0, stream>>>(enc16, wq16, wq_b, wk16, wk_b, q16, k16);
    attn_kernel<<<NBAT, 256, 0, stream>>>(q16, k16, out);
}
